// GATNetwork_89103391523368
// MI455X (gfx1250) — matrix operand order repaired and checked by reading, not compiled, not run on hardware
//
#include <hip/hip_runtime.h>
#include <hip/hip_fp16.h>
#include <math.h>

typedef __attribute__((ext_vector_type(16))) _Float16 v16h;
typedef __attribute__((ext_vector_type(8)))  _Float16 v8h;
typedef __attribute__((ext_vector_type(8)))  float    v8f;

#define IN_C      16
#define FDIM      20
#define HEADS1    4
#define HF1       80      // HEADS1*FDIM
#define EDIM      7
#define OUTC      3
#define NEG_SLOPE 0.2f

// ---------- helpers ----------
__device__ __forceinline__ unsigned f2ord(float f) {
    unsigned u = __float_as_uint(f);
    return (u & 0x80000000u) ? ~u : (u | 0x80000000u);
}
__device__ __forceinline__ float ord2f(unsigned u) {
    u = (u & 0x80000000u) ? (u & 0x7fffffffu) : ~u;
    return __uint_as_float(u);
}
__device__ __forceinline__ float elu1(float v) { return v > 0.f ? v : expm1f(v); }

// ---------- 1) degree + edge_attr sum per dst ----------
__global__ void k_deg_sum(const int* __restrict__ ei, const float* __restrict__ eattr,
                          float* __restrict__ deg, float* __restrict__ sumattr, int E) {
    int e = blockIdx.x * blockDim.x + threadIdx.x;
    if (e >= E) return;
    int d = ei[E + e];
    atomicAdd(&deg[d], 1.0f);
    const float* a = eattr + (size_t)e * EDIM;
    float* s = sumattr + (size_t)d * EDIM;
#pragma unroll
    for (int j = 0; j < EDIM; ++j) atomicAdd(&s[j], a[j]);
}

__global__ void k_mean(float* __restrict__ sumattr, const float* __restrict__ deg, int N) {
    int idx = blockIdx.x * blockDim.x + threadIdx.x;
    if (idx >= N * EDIM) return;
    int n = idx / EDIM;
    sumattr[idx] *= 1.0f / fmaxf(deg[n], 1.0f);
}

// ---------- 2) precompute Ce1[7][4], Ce2[7] ----------
__global__ void k_prep(const float* __restrict__ We1, const float* __restrict__ ae1,
                       const float* __restrict__ We2, const float* __restrict__ ae2,
                       float* __restrict__ Ce1, float* __restrict__ Ce2) {
    int t = threadIdx.x;
    if (t < EDIM * HEADS1) {
        int d = t / HEADS1, h = t % HEADS1;
        float acc = 0.f;
        for (int f = 0; f < FDIM; ++f) acc += We1[d * HF1 + h * FDIM + f] * ae1[h * FDIM + f];
        Ce1[d * HEADS1 + h] = acc;
    }
    if (t < EDIM) {
        float acc = 0.f;
        for (int f = 0; f < FDIM; ++f) acc += We2[t * FDIM + f] * ae2[f];
        Ce2[t] = acc;
    }
}

// ---------- 3) f32 -> padded f16 rows (for WMMA A operands) ----------
__global__ void k_pad16(const float* __restrict__ src, _Float16* __restrict__ dst,
                        int M, int K, int Kp) {
    int idx = blockIdx.x * blockDim.x + threadIdx.x;
    if (idx >= M * Kp) return;
    int m = idx / Kp, k = idx - m * Kp;
    dst[idx] = (k < K) ? (_Float16)src[(size_t)m * K + k] : (_Float16)0.f;
}

// ---------- 4) WMMA GEMM: C[M, CS] = Ah[M,Kp(f16)] @ B[K,NCOL(f32)] (zero-padded to NCP) ----
// 256 threads = 8 waves; each wave owns a 16-row M-tile, loops over padded N-tiles.
// A fragment: two contiguous v8h (b128) global loads per lane.
// B fragment: W transposed + zero-padded in LDS, two contiguous v8h ds loads per lane.
// Stores are unconditional (block-uniform fast path); pad columns store exact zeros.
template <int K, int Kp, int NCOL, int CS>
__global__ __launch_bounds__(256) void k_gemm_wmma(const _Float16* __restrict__ Ah,
                                                   const float* __restrict__ B,
                                                   float* __restrict__ C, int M) {
    constexpr int NCP = ((NCOL + 15) / 16) * 16;
    static_assert(CS >= NCP || CS == NCOL, "bad output stride");
    __shared__ _Float16 Wt[NCP * Kp];   // Wt[n][k], zero padded
    int t = threadIdx.x;
    for (int i = t; i < NCP * Kp; i += 256) {
        int n = i / Kp, k = i - n * Kp;
        Wt[i] = (n < NCOL && k < K) ? (_Float16)B[(size_t)k * NCOL + n] : (_Float16)0.f;
    }
    __syncthreads();

    int wave = t >> 5, lane = t & 31;
    int hi   = lane >> 4;                        // half-wave select
    int m0   = blockIdx.x * 128 + wave * 16;
    int mrow = m0 + (lane & 15);
    int ml   = (mrow < M) ? mrow : (M - 1);
    const _Float16* arow = Ah + (size_t)ml * Kp;
    bool blockFull = (blockIdx.x * 128 + 128) <= M;   // scalar (SGPR) condition

#pragma unroll
    for (int n0 = 0; n0 < NCP; n0 += 16) {
        int ncol = n0 + (lane & 15);
        const _Float16* bcol = &Wt[ncol * Kp];
        v8f c = {};
#pragma unroll
        for (int k0 = 0; k0 < K; k0 += 32) {
            v8h alo = *(const v8h*)(arow + k0 + hi * 8);
            v8h ahi = *(const v8h*)(arow + k0 + 16 + hi * 8);
            v8h blo = *(const v8h*)(bcol + k0 + hi * 8);
            v8h bhi = *(const v8h*)(bcol + k0 + 16 + hi * 8);
            v16h a = __builtin_shufflevector(alo, ahi, 0, 1, 2, 3, 4, 5, 6, 7,
                                             8, 9, 10, 11, 12, 13, 14, 15);
            v16h b = __builtin_shufflevector(blo, bhi, 0, 1, 2, 3, 4, 5, 6, 7,
                                             8, 9, 10, 11, 12, 13, 14, 15);
            c = __builtin_amdgcn_wmma_f32_16x16x32_f16(false, a, false, b, (short)0, c,
                                                       false, false);
        }
        if (blockFull) {                             // uniform branch, coalesced stores
#pragma unroll
            for (int r = 0; r < 8; ++r)
                C[(size_t)(m0 + r + hi * 8) * CS + ncol] = c[r];
        } else {                                     // last block only
#pragma unroll
            for (int r = 0; r < 8; ++r) {
                int row = m0 + r + hi * 8;
                if (row < M) C[(size_t)row * CS + ncol] = c[r];
            }
        }
    }
}

// ---------- 5) per-node attention scalars ----------
__global__ void k_attn_node(const float* __restrict__ xh, const float* __restrict__ a_s,
                            const float* __restrict__ a_d, float* __restrict__ asrc,
                            float* __restrict__ adst, int N, int H, int rowStride) {
    int idx = blockIdx.x * blockDim.x + threadIdx.x;
    if (idx >= N * H) return;
    int n = idx / H, h = idx % H;
    const float* row = xh + (size_t)n * rowStride + h * FDIM;
    float s = 0.f, d = 0.f;
#pragma unroll
    for (int f = 0; f < FDIM; ++f) {
        s += row[f] * a_s[h * FDIM + f];
        d += row[f] * a_d[h * FDIM + f];
    }
    asrc[idx] = s;
    adst[idx] = d;
}

// ---------- 6) edge logits + leaky relu + segment max ----------
__global__ void k_edge_logits(const int* __restrict__ ei, const float* __restrict__ eattr,
                              const float* __restrict__ meanattr,
                              const float* __restrict__ asrc, const float* __restrict__ adst,
                              const float* __restrict__ Ce, float* __restrict__ logits,
                              unsigned* __restrict__ maxb, int E, int N, int H) {
    int e = blockIdx.x * blockDim.x + threadIdx.x;
    int Ea = E + N;
    if (e >= Ea) return;
    int s, d;
    const float* ap;
    if (e < E) { s = ei[e]; d = ei[E + e]; ap = eattr + (size_t)e * EDIM; }
    else       { s = e - E; d = s;         ap = meanattr + (size_t)s * EDIM; }
    float at[EDIM];
#pragma unroll
    for (int j = 0; j < EDIM; ++j) at[j] = ap[j];
    for (int h = 0; h < H; ++h) {
        float ae = 0.f;
#pragma unroll
        for (int j = 0; j < EDIM; ++j) ae += at[j] * Ce[j * H + h];
        float l = asrc[s * H + h] + adst[d * H + h] + ae;
        l = l >= 0.f ? l : NEG_SLOPE * l;
        logits[(size_t)e * H + h] = l;
        atomicMax(&maxb[d * H + h], f2ord(l));
    }
}

// ---------- 7) exp(l - max) + segment denom ----------
__global__ void k_edge_exp(const int* __restrict__ ei, float* __restrict__ logits,
                           const unsigned* __restrict__ maxb, float* __restrict__ denom,
                           int E, int N, int H) {
    int e = blockIdx.x * blockDim.x + threadIdx.x;
    int Ea = E + N;
    if (e >= Ea) return;
    int d = (e < E) ? ei[E + e] : (e - E);
    for (int h = 0; h < H; ++h) {
        float l = logits[(size_t)e * H + h];
        float m = ord2f(maxb[d * H + h]);
        float ex = __expf(l - m);
        logits[(size_t)e * H + h] = ex;
        atomicAdd(&denom[d * H + h], ex);
    }
}

// ---------- 8) alpha-weighted message scatter ----------
__global__ void k_edge_scatter(const int* __restrict__ ei, const float* __restrict__ ex,
                               const float* __restrict__ denom, const float* __restrict__ xh,
                               float* __restrict__ out, int E, int N, int H,
                               int xRowStride, int oRowStride) {
    int idx = blockIdx.x * blockDim.x + threadIdx.x;
    int Ea = E + N;
    if (idx >= Ea * H) return;
    int e = idx / H, h = idx - e * H;
    int s, d;
    if (e < E) { s = ei[e]; d = ei[E + e]; }
    else       { s = e - E; d = s; }
    float alpha = ex[idx] / (denom[d * H + h] + 1e-16f);
    const float* xs = xh + (size_t)s * xRowStride + h * FDIM;
    float* od = out + (size_t)d * oRowStride + h * FDIM;
#pragma unroll
    for (int f = 0; f < FDIM; ++f) atomicAdd(&od[f], xs[f] * alpha);
}

// ---------- 9) bias + elu -> padded f16 rows for next GEMM ----------
__global__ void k_bias_elu16(const float* __restrict__ hacc, const float* __restrict__ b,
                             _Float16* __restrict__ h16, int N, int C, int Cp) {
    int idx = blockIdx.x * blockDim.x + threadIdx.x;
    if (idx >= N * Cp) return;
    int n = idx / Cp, c = idx - n * Cp;
    h16[idx] = (c < C) ? (_Float16)elu1(hacc[(size_t)n * C + c] + b[c]) : (_Float16)0.f;
}

// ---------- 10) conv2 finalize (bias+elu) + decoder MLP ----------
__global__ void k_final_decoder(const float* __restrict__ out2, const float* __restrict__ b2,
                                const float* __restrict__ D1w, const float* __restrict__ D1b,
                                const float* __restrict__ D2w, const float* __restrict__ D2b,
                                float* __restrict__ y, int N) {
    int n = blockIdx.x * blockDim.x + threadIdx.x;
    if (n >= N) return;
    float t[FDIM];
#pragma unroll
    for (int f = 0; f < FDIM; ++f) t[f] = elu1(out2[(size_t)n * FDIM + f] + b2[f]);
    float d1[10];
#pragma unroll
    for (int j = 0; j < 10; ++j) {
        float acc = D1b[j];
#pragma unroll
        for (int f = 0; f < FDIM; ++f) acc += t[f] * D1w[f * 10 + j];
        d1[j] = fmaxf(acc, 0.f);
    }
#pragma unroll
    for (int k = 0; k < OUTC; ++k) {
        float acc = D2b[k];
#pragma unroll
        for (int j = 0; j < 10; ++j) acc += d1[j] * D2w[j * OUTC + k];
        y[(size_t)n * OUTC + k] = acc;
    }
}

extern "C" void kernel_launch(void* const* d_in, const int* in_sizes, int n_in,
                              void* d_out, int out_size, void* d_ws, size_t ws_size,
                              hipStream_t stream) {
    const float* x     = (const float*)d_in[0];
    const int*   ei    = (const int*)  d_in[1];
    const float* eattr = (const float*)d_in[2];
    const float* W1    = (const float*)d_in[3];
    const float* as1   = (const float*)d_in[4];
    const float* ad1   = (const float*)d_in[5];
    const float* We1   = (const float*)d_in[6];
    const float* ae1   = (const float*)d_in[7];
    const float* b1    = (const float*)d_in[8];
    const float* W2    = (const float*)d_in[9];
    const float* as2   = (const float*)d_in[10];
    const float* ad2   = (const float*)d_in[11];
    const float* We2   = (const float*)d_in[12];
    const float* ae2   = (const float*)d_in[13];
    const float* b2    = (const float*)d_in[14];
    const float* D1w   = (const float*)d_in[15];
    const float* D1b   = (const float*)d_in[16];
    const float* D2w   = (const float*)d_in[17];
    const float* D2b   = (const float*)d_in[18];
    float* y = (float*)d_out;

    const int N  = in_sizes[0] / IN_C;
    const int E  = in_sizes[1] / 2;
    const int Ea = E + N;

    // workspace layout (float units; every region a multiple of 4 floats -> 16B aligned)
    float* w = (float*)d_ws;
    size_t o = 0;
    auto take = [&](size_t cnt) { float* p = w + o; o += cnt; return p; };
    float*     deg      = take(N);
    float*     meanattr = take((size_t)N * EDIM);
    float*     Ce1      = take(EDIM * HEADS1);           // 28
    float*     Ce2      = take(8);                       // 7 (+1 pad)
    float*     xh1      = take((size_t)N * HF1);
    float*     asrc1    = take((size_t)N * HEADS1);
    float*     adst1    = take((size_t)N * HEADS1);
    unsigned*  max1     = (unsigned*)take((size_t)N * HEADS1);
    float*     den1     = take((size_t)N * HEADS1);
    float*     logits   = take((size_t)Ea * HEADS1);     // reused by layer 2
    float*     h1acc    = take((size_t)N * HF1);
    float*     xh2      = take((size_t)N * 32);          // [N,32] padded (cols 20..31 zero)
    float*     asrc2    = take(N);
    float*     adst2    = take(N);
    unsigned*  max2     = (unsigned*)take(N);
    float*     den2     = take(N);
    float*     out2     = take((size_t)N * FDIM);
    _Float16*  x16      = (_Float16*)take((size_t)N * 32 / 2);   // [N,32] f16
    _Float16*  h16      = (_Float16*)take((size_t)N * 96 / 2);   // [N,96] f16
    (void)ws_size; (void)n_in; (void)out_size;

    const int B = 256;
    auto blks = [&](long long t) { return (int)((t + B - 1) / B); };

    // ---- self-loop mean edge_attr ----
    hipMemsetAsync(deg, 0, sizeof(float) * N, stream);
    hipMemsetAsync(meanattr, 0, sizeof(float) * (size_t)N * EDIM, stream);
    k_deg_sum<<<blks(E), B, 0, stream>>>(ei, eattr, deg, meanattr, E);
    k_mean<<<blks((size_t)N * EDIM), B, 0, stream>>>(meanattr, deg, N);
    k_prep<<<1, 32, 0, stream>>>(We1, ae1, We2, ae2, Ce1, Ce2);

    // ---- conv1: xh1 = x @ W1 (WMMA) ----
    k_pad16<<<blks((size_t)N * 32), B, 0, stream>>>(x, x16, N, IN_C, 32);
    k_gemm_wmma<IN_C, 32, HF1, HF1><<<(N + 127) / 128, 256, 0, stream>>>(x16, W1, xh1, N);
    k_attn_node<<<blks((size_t)N * HEADS1), B, 0, stream>>>(xh1, as1, ad1, asrc1, adst1,
                                                            N, HEADS1, HF1);
    hipMemsetAsync(max1, 0, sizeof(unsigned) * (size_t)N * HEADS1, stream);
    hipMemsetAsync(den1, 0, sizeof(float) * (size_t)N * HEADS1, stream);
    k_edge_logits<<<blks(Ea), B, 0, stream>>>(ei, eattr, meanattr, asrc1, adst1, Ce1,
                                              logits, max1, E, N, HEADS1);
    k_edge_exp<<<blks(Ea), B, 0, stream>>>(ei, logits, max1, den1, E, N, HEADS1);
    hipMemsetAsync(h1acc, 0, sizeof(float) * (size_t)N * HF1, stream);
    k_edge_scatter<<<blks((size_t)Ea * HEADS1), B, 0, stream>>>(ei, logits, den1, xh1, h1acc,
                                                                E, N, HEADS1, HF1, HF1);
    k_bias_elu16<<<blks((size_t)N * 96), B, 0, stream>>>(h1acc, b1, h16, N, HF1, 96);

    // ---- conv2: xh2 = h1 @ W2 (WMMA), output padded to stride 32 ----
    k_gemm_wmma<HF1, 96, FDIM, 32><<<(N + 127) / 128, 256, 0, stream>>>(h16, W2, xh2, N);
    k_attn_node<<<blks(N), B, 0, stream>>>(xh2, as2, ad2, asrc2, adst2, N, 1, 32);
    hipMemsetAsync(max2, 0, sizeof(unsigned) * N, stream);
    hipMemsetAsync(den2, 0, sizeof(float) * N, stream);
    k_edge_logits<<<blks(Ea), B, 0, stream>>>(ei, eattr, meanattr, asrc2, adst2, Ce2,
                                              logits, max2, E, N, 1);
    k_edge_exp<<<blks(Ea), B, 0, stream>>>(ei, logits, max2, den2, E, N, 1);
    hipMemsetAsync(out2, 0, sizeof(float) * (size_t)N * FDIM, stream);
    k_edge_scatter<<<blks(Ea), B, 0, stream>>>(ei, logits, den2, xh2, out2, E, N, 1, 32, FDIM);

    // ---- finalize + decoder ----
    k_final_decoder<<<blks(N), B, 0, stream>>>(out2, b2, D1w, D1b, D2w, D2b, y, N);
}
